// GCN_75557064671667
// MI455X (gfx1250) — matrix operand order repaired and checked by reading, not compiled, not run on hardware
//
#include <hip/hip_runtime.h>

// ---------------------------------------------------------------------------
// GCN layer on MI455X (gfx1250, wave32):
//   h   = mean(n_feat, axis=1)           [2048,2048]  (memory-bound: 256 MiB)
//   out = h @ W^T + b                    [2048,256]   (WMMA f16, fp32 accum)
//
// HBM floor = 256MiB / 23.3 TB/s ~= 11.5us (pass 1). The WMMA GEMM is only
// 2.1 GFLOP; split-K by 8 (4096 waves, 8 K-steps each) keeps the machine full
// and hides L2 latency, with a deterministic reduce (+bias) afterwards.
// Extra partial traffic: 32 MiB ~ 1.4us against the 23.3 TB/s budget.
// ---------------------------------------------------------------------------

typedef __attribute__((ext_vector_type(4)))  float    v4f;
typedef __attribute__((ext_vector_type(8)))  float    v8f;
typedef __attribute__((ext_vector_type(4)))  _Float16 v4h;
typedef __attribute__((ext_vector_type(8)))  _Float16 v8h;
typedef __attribute__((ext_vector_type(16))) _Float16 v16h;

#define NN     2048   // nodes / in_features
#define KK     16     // replica axis (mean)
#define OUT    256    // out_features
#define SPLITK 8      // f-dimension split for the GEMM
#define FCHUNK (NN / SPLITK)   // 256 f-values per split

// ---------------------------------------------------------------------------
// Pass 1: h[n,f] = (1/16) * sum_k n_feat[n,k,f], stored as f16.
// 1,048,576 threads, one float4 of h each; 16 strided (8KB apart) v4f
// non-temporal loads (single-use stream, keep out of L2), fully unrolled.
// ---------------------------------------------------------------------------
__global__ void gcn_mean_kernel(const float* __restrict__ nf,
                                _Float16* __restrict__ h) {
    const int t  = blockIdx.x * blockDim.x + threadIdx.x; // 0 .. 2048*512-1
    const int f4 = t & 511;
    const int n  = t >> 9;
    const float* base = nf + (size_t)n * (KK * NN) + (size_t)f4 * 4;

    v4f s = (v4f)(0.0f);
#pragma unroll
    for (int k = 0; k < KK; ++k) {
        v4f v = __builtin_nontemporal_load((const v4f*)(base + (size_t)k * NN));
        s += v;
    }
    s *= 0.0625f;

    v4h o;
    o[0] = (_Float16)s[0];
    o[1] = (_Float16)s[1];
    o[2] = (_Float16)s[2];
    o[3] = (_Float16)s[3];
    *(v4h*)(h + (size_t)n * NN + (size_t)f4 * 4) = o;
}

// ---------------------------------------------------------------------------
// Pass 2: W (fp32, [OUT,NN] row-major) -> f16.
// ---------------------------------------------------------------------------
__global__ void gcn_wconv_kernel(const float* __restrict__ W,
                                 _Float16* __restrict__ Wh) {
    const int t = blockIdx.x * blockDim.x + threadIdx.x;  // 0 .. OUT*NN/4-1
    v4f w = *(const v4f*)(W + (size_t)t * 4);
    v4h o;
    o[0] = (_Float16)w[0];
    o[1] = (_Float16)w[1];
    o[2] = (_Float16)w[2];
    o[3] = (_Float16)w[3];
    *(v4h*)(Wh + (size_t)t * 4) = o;
}

// ---------------------------------------------------------------------------
// Pass 3: split-K WMMA GEMM partials.
//   partial[s][n][o] = sum_{f in chunk s} h[n,f] * Wh[o,f]
// One wave -> 16 rows x 64 cols over 8 K-steps of 32.
// 512 strips (128 rowT x 4 colG) x 8 splits = 4096 waves = 512 blocks x 8.
// Consecutive waves in a block share the same split and neighbouring strips,
// so A rows are reused via L1 across the 4 col-groups of a row tile.
//
// Fragment layouts as loaded here (wave32; contiguous k, same order in A and B):
//   A 16x32 f16 : grp0 (lanes 0-15): M=lane,   K = 0..15
//                 grp1 (lanes16-31): M=lane-16, K = 16..31
//   B 32x16 f16 : lane = N column; grp0 holds K=0..15, grp1 K=16..31
//   C/D 16x16 f32: VGPR r -> M = r + 8*grp, N = lane & 15
//
// No divergence anywhere: EXEC is all-1s as WMMA requires; the prefetch is
// unconditional (speculative, dropped on fault; one-past-end address stays
// inside the workspace) so the inner loop keeps pure scalar control flow.
// ---------------------------------------------------------------------------
__global__ void gcn_wmma_gemm_kernel(const _Float16* __restrict__ h,
                                     const _Float16* __restrict__ Wh,
                                     float* __restrict__ partial) {
    const int tid   = threadIdx.x;
    const int wave  = tid >> 5;
    const int lane  = tid & 31;
    const int gw    = blockIdx.x * 8 + wave;   // 0..4095
    const int strip = gw & 511;                // 0..511
    const int split = gw >> 9;                 // 0..7
    const int rowT  = strip >> 2;              // 0..127
    const int colG  = strip & 3;               // 0..3
    const int nBase = rowT * 16;
    const int oBase = colG * 64;
    const int fBase = split * FCHUNK;
    const int grp   = lane >> 4;               // 0 or 1
    const int nl    = lane & 15;

    const _Float16* aRow  = h  + (size_t)(nBase + nl) * NN;
    const _Float16* bRow0 = Wh + (size_t)(oBase +  0 + nl) * NN + 16 * grp;
    const _Float16* bRow1 = Wh + (size_t)(oBase + 16 + nl) * NN + 16 * grp;
    const _Float16* bRow2 = Wh + (size_t)(oBase + 32 + nl) * NN + 16 * grp;
    const _Float16* bRow3 = Wh + (size_t)(oBase + 48 + nl) * NN + 16 * grp;

    v8f acc0 = (v8f)(0.0f), acc1 = (v8f)(0.0f);
    v8f acc2 = (v8f)(0.0f), acc3 = (v8f)(0.0f);

#pragma unroll 2
    for (int f = fBase; f < fBase + FCHUNK; f += 32) {
        // A fragment: two 16B chunks (K 0-15 for grp0, K 16-31 for grp1)
        v8h alo = *(const v8h*)(aRow + f +     16 * grp);
        v8h ahi = *(const v8h*)(aRow + f + 8 + 16 * grp);
        v16h a;
#pragma unroll
        for (int i = 0; i < 8; ++i) { a[i] = alo[i]; a[8 + i] = ahi[i]; }

        // speculative next-chunk prefetch (global_prefetch_b8); no guard so
        // the loop keeps scalar control flow and EXEC stays all-1s
        __builtin_prefetch(aRow + f + 32, 0, 3);

        // B fragments: 16 contiguous halves (32B) per lane
        v16h b0 = *(const v16h*)(bRow0 + f);
        v16h b1 = *(const v16h*)(bRow1 + f);
        v16h b2 = *(const v16h*)(bRow2 + f);
        v16h b3 = *(const v16h*)(bRow3 + f);

        acc0 = __builtin_amdgcn_wmma_f32_16x16x32_f16(false, a, false, b0,
                                                      (short)0, acc0, false, false);
        acc1 = __builtin_amdgcn_wmma_f32_16x16x32_f16(false, a, false, b1,
                                                      (short)0, acc1, false, false);
        acc2 = __builtin_amdgcn_wmma_f32_16x16x32_f16(false, a, false, b2,
                                                      (short)0, acc2, false, false);
        acc3 = __builtin_amdgcn_wmma_f32_16x16x32_f16(false, a, false, b3,
                                                      (short)0, acc3, false, false);
    }

    // Store partials (no bias here; reduce kernel adds it deterministically)
    float* pslab = partial + (size_t)split * (NN * OUT);
#pragma unroll
    for (int r = 0; r < 8; ++r) {
        const int m = nBase + grp * 8 + r;
        float* prow = pslab + (size_t)m * OUT + oBase + nl;
        prow[ 0] = acc0[r];
        prow[16] = acc1[r];
        prow[32] = acc2[r];
        prow[48] = acc3[r];
    }
}

// ---------------------------------------------------------------------------
// Pass 4: out[n,o] = bias[o] + sum_s partial[s][n][o]   (fixed order, exact
// same result every call). 131072 threads, one float4 each; partials are
// L2-resident (16 MiB slab).
// ---------------------------------------------------------------------------
__global__ void gcn_reduce_kernel(const float* __restrict__ partial,
                                  const float* __restrict__ bias,
                                  float* __restrict__ out) {
    const int t  = blockIdx.x * blockDim.x + threadIdx.x; // 0 .. 2048*64-1
    const int o4 = t & 63;                                // float4 in out row
    const int n  = t >> 6;
    const size_t off = (size_t)n * OUT + (size_t)o4 * 4;

    v4f s = *(const v4f*)(bias + (size_t)o4 * 4);
#pragma unroll
    for (int sp = 0; sp < SPLITK; ++sp)
        s += *(const v4f*)(partial + (size_t)sp * (NN * OUT) + off);

    *(v4f*)(out + off) = s;
}

extern "C" void kernel_launch(void* const* d_in, const int* in_sizes, int n_in,
                              void* d_out, int out_size, void* d_ws, size_t ws_size,
                              hipStream_t stream) {
    const float* nf = (const float*)d_in[0];   // [2048,16,2048] fp32
    const float* W  = (const float*)d_in[1];   // [256,2048]     fp32
    const float* b  = (const float*)d_in[2];   // [256]          fp32
    float* out      = (float*)d_out;           // [2048,256]     fp32

    // workspace: h_f16 (8 MiB) | W_f16 (1 MiB) | partials (16 MiB)
    _Float16* h   = (_Float16*)d_ws;
    _Float16* Wh  = (_Float16*)((char*)d_ws + (size_t)NN * NN * sizeof(_Float16));
    float* part   = (float*)((char*)d_ws + (size_t)NN * NN * sizeof(_Float16)
                                         + (size_t)OUT * NN * sizeof(_Float16));

    // Pass 1: mean over K -> f16   (2048*2048/4 threads)
    gcn_mean_kernel<<<(NN * NN / 4) / 256, 256, 0, stream>>>(nf, h);

    // Pass 2: W fp32 -> f16        (256*2048/4 threads)
    gcn_wconv_kernel<<<(OUT * NN / 4) / 256, 256, 0, stream>>>(W, Wh);

    // Pass 3: split-K WMMA partials: 4096 waves = 512 blocks x 256 threads
    gcn_wmma_gemm_kernel<<<512, 256, 0, stream>>>(h, Wh, part);

    // Pass 4: deterministic reduce + bias (2048*256/4 threads)
    gcn_reduce_kernel<<<(NN * OUT / 4) / 256, 256, 0, stream>>>(part, b, out);
}
